// Router_69432441307499
// MI455X (gfx1250) — matrix operand order repaired and checked by reading, not compiled, not run on hardware
//
#include <hip/hip_runtime.h>
#include <hip/hip_bf16.h>

#define NUM_EXPERTS 64
#define SEQ 4096
#define BATCH 8
#define HID 4096
#define CAP 128
#define TOK_PER_BLOCK 128

typedef __attribute__((ext_vector_type(16))) __bf16 v16bf;
typedef __attribute__((ext_vector_type(8)))  __bf16 v8bf;
typedef __attribute__((ext_vector_type(8)))  float  v8f;
typedef __attribute__((ext_vector_type(4)))  float  v4f;
typedef __attribute__((ext_vector_type(4)))  unsigned int v4u;
typedef __attribute__((ext_vector_type(8)))  int    v8i;
typedef __attribute__((ext_vector_type(4)))  int    v4i;

#if defined(__gfx1250__) && __has_builtin(__builtin_amdgcn_tensor_load_to_lds) && \
    __has_builtin(__builtin_amdgcn_s_wait_tensorcnt)
#define USE_TDM 1
#else
#define USE_TDM 0
#endif

// ---------------- Kernel 0: fp32 -> bf16 weight conversion ----------------
__global__ void cvt_weight_kernel(const float* __restrict__ w,
                                  __bf16* __restrict__ wb, int n) {
  int i = blockIdx.x * blockDim.x + threadIdx.x;
  if (i < n) wb[i] = (__bf16)w[i];
}

#if USE_TDM
// Issue a TDM load of a [64 experts x 32 K] bf16 tile (4 KB) into LDS.
// D# bitfields per CDNA5 ISA:
//   group0: count=1 | lds_addr | global_addr[56:0] | type=2
//   group1: data_size=2B, tensor_dim0=HID, tensor_dim1=E, tile_dim0=32,
//           tile_dim1=E, tensor_dim0_stride=HID
// This toolchain exposes the 6-arg builtin:
//   (uint32x4 g0, int32x8 g1, int32x4, int32x4, int32x8, i32 cpol)
__device__ __forceinline__ void tdm_load_b_chunk(const __bf16* gsrc,
                                                 unsigned lds_off) {
  unsigned long long ga = (unsigned long long)(uintptr_t)gsrc;
  v4u g0;
  g0[0] = 1u;                                        // count=1, user mode
  g0[1] = lds_off;                                   // lds_addr (bytes)
  g0[2] = (unsigned)(ga & 0xffffffffu);              // global_addr[31:0]
  g0[3] = (unsigned)((ga >> 32) & 0x01ffffffu)       // global_addr[56:32]
        | (2u << 30);                                // type=2 ("image")
  v8i g1;
  g1[0] = (int)(1u << 16);                           // data_size=1 (2 bytes)
  g1[1] = (int)(((unsigned)(HID & 0xffff)) << 16);   // tensor_dim0[15:0] @ [63:48]
  g1[2] = (int)(((unsigned)(NUM_EXPERTS & 0xffff)) << 16
        | ((HID >> 16) & 0xffff));                   // dim0[31:16] | dim1[15:0]
  g1[3] = (int)(32u << 16);                          // tile_dim0=32 @ [127:112]
  g1[4] = NUM_EXPERTS;                               // tile_dim1=64 @ [143:128]
  g1[5] = HID;                                       // tensor_dim0_stride[31:0]
  g1[6] = 0;
  g1[7] = 0;
  v4i z4 = {0, 0, 0, 0};                             // 2D: groups 2/3 unused
  v8i z8 = {0, 0, 0, 0, 0, 0, 0, 0};
  __builtin_amdgcn_tensor_load_to_lds(g0, g1, z4, z4, z8, 0);
}
#endif

// ---------------- Kernel 1: WMMA router GEMM + softmax + top2 -------------
// Block = 256 threads = 8 wave32. Each wave computes a 16-token x 64-expert
// logits strip via V_WMMA_F32_16X16X32_BF16. B chunks (64x32 bf16, 4 KB) are
// staged in LDS by the Tensor Data Mover (double-buffered, TENSORcnt).
__global__ __launch_bounds__(256) void router_wmma_kernel(
    const float*  __restrict__ hs,       // [B*S, H] fp32
    const __bf16* __restrict__ wb,       // [E, H]   bf16
    float* __restrict__ out,             // d_out: dispatch|combine|probs|aux|z
    int*   __restrict__ top_idx,         // [B*S, 2]
    float* __restrict__ top_w,           // [B*S, 2]
    float* __restrict__ sump2,           // [B*S]
    float* __restrict__ lse2)            // [B*S]
{
  const long BSE = (long)BATCH * SEQ * NUM_EXPERTS;
  float* dispatch = out;
  float* combine  = out + BSE;
  float* probs    = out + 2 * BSE;

  __shared__ float lds_logits[8 * 16 * NUM_EXPERTS];            // 32 KB
  __shared__ __align__(16) __bf16 bstage[2][NUM_EXPERTS * 32];  // 2 x 4 KB

  const int tid  = threadIdx.x;
  const int wave = tid >> 5;
  const int lane = tid & 31;
  const int hi   = lane >> 4;       // lane half (ISA A/B layout split)
  const int ln   = lane & 15;

  const long tokBase = (long)blockIdx.x * TOK_PER_BLOCK;
  const long waveTok = tokBase + (long)wave * 16;

  // Zero-init this block's slice of dispatch/combine (scan kernel fills later).
  for (int i = tid; i < TOK_PER_BLOCK * NUM_EXPERTS; i += 256) {
    long g = tokBase * NUM_EXPERTS + i;
    dispatch[g] = 0.0f;
    combine[g]  = 0.0f;
  }

  v8f acc0 = {}, acc1 = {}, acc2 = {}, acc3 = {};

  // A-matrix row for this lane: token waveTok + ln, M = ln.
  const float* aRow = hs + (waveTok + ln) * (long)HID;

#if USE_TDM
  // Prologue: wave 0 DMAs chunk 0 into buffer 0.
  if (tid < 32) tdm_load_b_chunk(wb, (unsigned)(uintptr_t)&bstage[0][0]);
#endif

  for (int kb = 0; kb < HID; kb += 32) {
    const int cur = (kb >> 5) & 1;
#if USE_TDM
    const bool more = (kb + 32) < HID;
    if (tid < 32) {
      if (more) {
        tdm_load_b_chunk(wb + kb + 32, (unsigned)(uintptr_t)&bstage[cur ^ 1][0]);
        __builtin_amdgcn_s_wait_tensorcnt(1);  // chunk `cur` landed
      } else {
        __builtin_amdgcn_s_wait_tensorcnt(0);
      }
    }
    __syncthreads();   // B chunk `cur` visible to all waves
#else
    // Fallback: cooperative staged copy (256 threads x 16B = 4 KB).
    {
      const int e = tid >> 2, part = tid & 3;
      v8bf t = *(const v8bf*)(wb + (size_t)e * HID + kb + part * 8);
      *(v8bf*)&bstage[cur][e * 32 + part * 8] = t;
    }
    __syncthreads();
#endif

    // --- A tile 16x32 bf16, ISA layout:
    // elem j<8 : K = kb + hi*8 + j ; elem j>=8 : K = kb + 16 + hi*8 + (j-8)
    v4f a0 = *(const v4f*)(aRow + kb + hi * 8);
    v4f a1 = *(const v4f*)(aRow + kb + hi * 8 + 4);
    v4f a2 = *(const v4f*)(aRow + kb + 16 + hi * 8);
    v4f a3 = *(const v4f*)(aRow + kb + 16 + hi * 8 + 4);
    v16bf a;
#pragma unroll
    for (int j = 0; j < 4; ++j) {
      a[j]      = (__bf16)a0[j];
      a[4 + j]  = (__bf16)a1[j];
      a[8 + j]  = (__bf16)a2[j];
      a[12 + j] = (__bf16)a3[j];
    }

    // --- B tiles from LDS: lane -> N = ln, elem j<8 -> K = kb + hi*8 + j ; j>=8 -> K = kb + 16 + hi*8 + (j-8).
    const __bf16* bb = &bstage[cur][0];
#pragma unroll
    for (int nt = 0; nt < 4; ++nt) {
      const __bf16* bp = bb + (nt * 16 + ln) * 32;
      v8bf blo = *(const v8bf*)(bp + hi * 8);
      v8bf bhi = *(const v8bf*)(bp + 16 + hi * 8);
      v16bf b;
#pragma unroll
      for (int j = 0; j < 8; ++j) { b[j] = blo[j]; b[8 + j] = bhi[j]; }
      if (nt == 0)
        acc0 = __builtin_amdgcn_wmma_f32_16x16x32_bf16(false, a, false, b, (short)0, acc0, false, false);
      else if (nt == 1)
        acc1 = __builtin_amdgcn_wmma_f32_16x16x32_bf16(false, a, false, b, (short)0, acc1, false, false);
      else if (nt == 2)
        acc2 = __builtin_amdgcn_wmma_f32_16x16x32_bf16(false, a, false, b, (short)0, acc2, false, false);
      else
        acc3 = __builtin_amdgcn_wmma_f32_16x16x32_bf16(false, a, false, b, (short)0, acc3, false, false);
    }
    __syncthreads();   // protect bstage[cur] before it is refilled
  }

  // Spill logits to LDS. C/D layout: lane -> N = ln, VGPR r -> M = r + 8*hi.
  float* myLds = lds_logits + wave * 16 * NUM_EXPERTS;
#pragma unroll
  for (int r = 0; r < 8; ++r) {
    int m = r + 8 * hi;
    myLds[m * NUM_EXPERTS +  0 + ln] = acc0[r];
    myLds[m * NUM_EXPERTS + 16 + ln] = acc1[r];
    myLds[m * NUM_EXPERTS + 32 + ln] = acc2[r];
    myLds[m * NUM_EXPERTS + 48 + ln] = acc3[r];
  }
  __syncthreads();

  // Softmax / top-2 / loss terms: lanes 0..15 each own one token of the wave.
  if (hi == 0) {
    const float* lrow = myLds + ln * NUM_EXPERTS;
    float mx = -INFINITY;
#pragma unroll 8
    for (int e = 0; e < NUM_EXPERTS; ++e) mx = fmaxf(mx, lrow[e]);
    float sum = 0.0f;
#pragma unroll 8
    for (int e = 0; e < NUM_EXPERTS; ++e) sum += __expf(lrow[e] - mx);
    float inv = 1.0f / sum;

    long g = waveTok + ln;
    float* prow = probs + g * NUM_EXPERTS;
    float p0 = -1.0f, p1 = -1.0f;
    int   i0 = 0,     i1 = 0;
    float sp2 = 0.0f;
    for (int e = 0; e < NUM_EXPERTS; ++e) {
      float p = __expf(lrow[e] - mx) * inv;
      prow[e] = p;
      sp2 += p * p;
      if (p > p0)      { p1 = p0; i1 = i0; p0 = p; i0 = e; }
      else if (p > p1) { p1 = p;  i1 = e; }
    }
    float wsum = p0 + p1;
    top_idx[g * 2 + 0] = i0;
    top_idx[g * 2 + 1] = i1;
    top_w[g * 2 + 0] = p0 / wsum;
    top_w[g * 2 + 1] = p1 / wsum;
    sump2[g] = sp2;
    float lse = mx + __logf(sum);
    lse2[g] = lse * lse;
  }
}

// ---------------- Kernel 2: ordered capacity-limited dispatch -------------
__global__ void dispatch_scan_kernel(const int*   __restrict__ top_idx,
                                     const float* __restrict__ top_w,
                                     float* __restrict__ out)
{
  const long BSE = (long)BATCH * SEQ * NUM_EXPERTS;
  float* dispatch = out;
  float* combine  = out + BSE;
  const int b = blockIdx.x;
  const int e = threadIdx.x;
  const long base = (long)b * SEQ;
  int cnt = 0;
  for (int k = 0; k < 2; ++k) {
    for (int s = 0; s < SEQ; ++s) {
      long g = base + s;
      int idx = top_idx[g * 2 + k];
      if (idx == e) {
        if (cnt < CAP) {
          long o = g * NUM_EXPERTS + e;
          dispatch[o] = 1.0f;
          combine[o]  = top_w[g * 2 + k];
        }
        ++cnt;
      }
    }
  }
}

// ---------------- Kernel 3: deterministic loss reduction ------------------
__global__ void loss_reduce_kernel(const float* __restrict__ sump2,
                                   const float* __restrict__ lse2,
                                   float* __restrict__ out)
{
  const int N = BATCH * SEQ;
  __shared__ float sa[256], sb[256];
  int t = threadIdx.x;
  float s1 = 0.0f, s2 = 0.0f;
  for (int i = t; i < N; i += 256) { s1 += sump2[i]; s2 += lse2[i]; }
  sa[t] = s1; sb[t] = s2;
  __syncthreads();
  for (int w = 128; w > 0; w >>= 1) {
    if (t < w) { sa[t] += sa[t + w]; sb[t] += sb[t + w]; }
    __syncthreads();
  }
  if (t == 0) {
    const long BSE = (long)BATCH * SEQ * NUM_EXPERTS;
    out[3 * BSE]     = sa[0] / (float)N * (float)NUM_EXPERTS;  // aux_loss
    out[3 * BSE + 1] = sb[0] / (float)N;                       // z_loss
  }
}

extern "C" void kernel_launch(void* const* d_in, const int* in_sizes, int n_in,
                              void* d_out, int out_size, void* d_ws, size_t ws_size,
                              hipStream_t stream) {
  const float* hs = (const float*)d_in[0];   // [8,4096,4096]
  const float* rw = (const float*)d_in[1];   // [64,4096]
  float* out = (float*)d_out;
  char* ws = (char*)d_ws;

  const long NT = (long)BATCH * SEQ;         // 32768 tokens

  size_t off = 0;
  __bf16* wb = (__bf16*)(ws + off);          off += (size_t)NUM_EXPERTS * HID * sizeof(__bf16);
  off = (off + 255) & ~(size_t)255;
  int*   top_idx = (int*)(ws + off);         off += (size_t)NT * 2 * sizeof(int);
  float* top_w   = (float*)(ws + off);       off += (size_t)NT * 2 * sizeof(float);
  float* sump2   = (float*)(ws + off);       off += (size_t)NT * sizeof(float);
  float* lse2    = (float*)(ws + off);

  cvt_weight_kernel<<<(NUM_EXPERTS * HID + 255) / 256, 256, 0, stream>>>(
      rw, wb, NUM_EXPERTS * HID);

  router_wmma_kernel<<<NT / TOK_PER_BLOCK, 256, 0, stream>>>(
      hs, wb, out, top_idx, top_w, sump2, lse2);

  dispatch_scan_kernel<<<BATCH, NUM_EXPERTS, 0, stream>>>(top_idx, top_w, out);

  loss_reduce_kernel<<<1, 256, 0, stream>>>(sump2, lse2, out);
}
